// GCN7_71511205478660
// MI455X (gfx1250) — matrix operand order repaired and checked by reading, not compiled, not run on hardware
//
#include <hip/hip_runtime.h>
#include <cstdint>
#include <cstddef>

// ---------------------------------------------------------------------------
// GCN7 on gfx1250 (MI455X): FP16 activations + v_wmma_f32_16x16x32_f16 GEMMs,
// FP32 atomic edge aggregation, degree norm folded into f32->f16 conversion.
// ---------------------------------------------------------------------------

#define NNODES 100000
#define NEDGES 800000
#define NGRAPH 100
#define LD 128              // padded feature stride (halves), uniform all layers
#define KTILES 4            // 4 x 32 = 128 K
#define NTILES 8            // 8 x 16 = 128 N (zero-padded weight panels)
#define TILE_HALVES 512     // 32 lanes * 16 halves per 32x16 B tile
#define PACKED_HALVES (KTILES * NTILES * TILE_HALVES)   // 16384 halves / matrix

typedef _Float16 f16;
typedef __attribute__((ext_vector_type(16))) _Float16 v16h;
typedef __attribute__((ext_vector_type(8)))  _Float16 v8h;
typedef __attribute__((ext_vector_type(8)))  float    v8f;

// ---------------- degree kernels -------------------------------------------

__global__ void deg_count(const int* __restrict__ src, const int* __restrict__ dst,
                          float* __restrict__ inDeg, float* __restrict__ outDeg) {
  int e = blockIdx.x * blockDim.x + threadIdx.x;
  if (e >= NEDGES) return;
  atomicAdd(&inDeg[dst[e]], 1.0f);
  atomicAdd(&outDeg[src[e]], 1.0f);
}

__global__ void deg_final(const float* __restrict__ inDeg, const float* __restrict__ outDeg,
                          float* __restrict__ invIn, float* __restrict__ inIsq,
                          float* __restrict__ outIsq) {
  int n = blockIdx.x * blockDim.x + threadIdx.x;
  if (n >= NNODES) return;
  float di = fmaxf(inDeg[n], 1.0f);
  invIn[n] = 1.0f / di;
  inIsq[n] = rsqrtf(di);
  float dq = fmaxf(outDeg[n], 1.0f);
  outIsq[n] = rsqrtf(dq);
}

// ---------------- conversions ----------------------------------------------

__global__ void cvt_f32_f16(const float* __restrict__ X, f16* __restrict__ O, int n) {
  int i = blockIdx.x * blockDim.x + threadIdx.x;
  if (i < n) O[i] = (f16)X[i];
}

// aggh[n][c] = f16( aggf[n][c] * rowScale[n] )  (1/deg for SAGE, in_isqrt for GCN)
__global__ void cvt_agg(const float* __restrict__ A, const float* __restrict__ rowScale,
                        f16* __restrict__ O) {
  int i = blockIdx.x * blockDim.x + threadIdx.x;
  if (i < NNODES * LD) O[i] = (f16)(A[i] * rowScale[i >> 7]);
}

// ---------------- weight packing into WMMA B layout ------------------------
// B (32x16 f16 per tile): lane half h, half j holds K = kt*32 + 8h + (j&7)
// + 16*(j>>3); column n = nt*16 + lane%16. Zero-pad beyond (K, Nout).

__global__ void pack_w(const float* __restrict__ W, int K, int Nout,
                       f16* __restrict__ P) {
  int idx = blockIdx.x * blockDim.x + threadIdx.x;
  if (idx >= PACKED_HALVES) return;
  int t = idx >> 9;               // tile index kt*NTILES + nt
  int r = idx & 511;
  int kt = t >> 3, nt = t & 7;
  int l = r >> 4, j = r & 15;     // lane, half-within-lane
  int k = kt * 32 + ((l >> 4) << 3) + (j & 7) + ((j >> 3) << 4);
  int n = nt * 16 + (l & 15);
  float v = (k < K && n < Nout) ? W[k * Nout + n] : 0.0f;
  P[idx] = (f16)v;
}

// ---------------- edge aggregation (scatter-add) ---------------------------
// 32 lanes per edge, 4 channels per lane. srcScale==nullptr -> SAGE (raw sum),
// otherwise GCN gather scale h[src]*out_isqrt[src].

__global__ void agg_scatter(const f16* __restrict__ H, const int* __restrict__ src,
                            const int* __restrict__ dst, const float* __restrict__ srcScale,
                            float* __restrict__ Agg) {
  int idx = blockIdx.x * blockDim.x + threadIdx.x;
  int e = idx >> 5;
  if (e >= NEDGES) return;
  int lane = idx & 31;
  int s = src[e], d = dst[e];
  float sc = (srcScale != nullptr) ? srcScale[s] : 1.0f;
  const f16* hp = H + (size_t)s * LD + lane * 4;
  float* ap = Agg + (size_t)d * LD + lane * 4;
#pragma unroll
  for (int i = 0; i < 4; ++i)
    atomicAdd(ap + i, (float)hp[i] * sc);
}

// ---------------- WMMA GEMM -------------------------------------------------
// Out[N x 128] = relu( A1@B1 (+ A2@B2) + bias ), all operands padded to 128.
// One wave per 16x16 output tile; grid is exact so EXEC is all-ones.

__device__ __forceinline__ v16h load_a_tile(const f16* p) {
  // 16-bit A layout: halves 0-7 = K chunk at p, halves 8-15 = chunk at p+16
  v8h lo = *(const v8h*)p;
  v8h hi = *(const v8h*)(p + 16);
  v16h r;
#pragma unroll
  for (int i = 0; i < 8; ++i) { r[i] = lo[i]; r[i + 8] = hi[i]; }
  return r;
}

__global__ __launch_bounds__(256)
void gemm_wmma(const f16* __restrict__ A1, const f16* __restrict__ B1,
               const f16* __restrict__ A2, const f16* __restrict__ B2,
               const float* __restrict__ bias, int Nout, int doRelu,
               f16* __restrict__ Out) {
  const int wave = blockIdx.x * (blockDim.x >> 5) + (threadIdx.x >> 5);
  const int lane = threadIdx.x & 31;
  const int mt = wave >> 3;        // M tile (16 rows)
  const int nt = wave & 7;         // N tile (16 cols)
  const int hs = lane >> 4;        // K-chunk selector
  const int ln = lane & 15;
  const size_t rowA = (size_t)(mt * 16 + ln) * LD;

  v8f acc = {};
#pragma unroll
  for (int kt = 0; kt < KTILES; ++kt) {
    v16h a = load_a_tile(A1 + rowA + kt * 32 + hs * 8);
    v16h b = *(const v16h*)(B1 + (size_t)(kt * NTILES + nt) * TILE_HALVES + lane * 16);
    acc = __builtin_amdgcn_wmma_f32_16x16x32_f16(false, a, false, b,
                                                 (short)0, acc, false, false);
  }
  if (A2 != nullptr) {
#pragma unroll
    for (int kt = 0; kt < KTILES; ++kt) {
      v16h a = load_a_tile(A2 + rowA + kt * 32 + hs * 8);
      v16h b = *(const v16h*)(B2 + (size_t)(kt * NTILES + nt) * TILE_HALVES + lane * 16);
      acc = __builtin_amdgcn_wmma_f32_16x16x32_f16(false, a, false, b,
                                                   (short)0, acc, false, false);
    }
  }

  const int n = nt * 16 + ln;
  const float bv = (n < Nout) ? bias[n] : 0.0f;  // pad cols: 0 + 0 -> stored 0
#pragma unroll
  for (int r = 0; r < 8; ++r) {
    int row = mt * 16 + hs * 8 + r;              // C/D layout: M = r + 8*hs
    float v = acc[r] + bv;
    if (doRelu) v = fmaxf(v, 0.0f);
    Out[(size_t)row * LD + n] = (f16)v;
  }
}

// ---------------- graph mean pooling ---------------------------------------

__global__ void pool_sum(const f16* __restrict__ H, const int* __restrict__ gid,
                         float* __restrict__ out, float* __restrict__ cnt) {
  int n = blockIdx.x * blockDim.x + threadIdx.x;
  if (n >= NNODES) return;
  int g = gid[n];
  const f16* hp = H + (size_t)n * LD;
#pragma unroll
  for (int c = 0; c < 5; ++c)
    atomicAdd(&out[g * 5 + c], (float)hp[c]);
  atomicAdd(&cnt[g], 1.0f);
}

__global__ void pool_div(float* __restrict__ out, const float* __restrict__ cnt) {
  int i = blockIdx.x * blockDim.x + threadIdx.x;
  if (i < NGRAPH * 5) out[i] /= fmaxf(cnt[i / 5], 1.0f);
}

// ---------------------------------------------------------------------------

extern "C" void kernel_launch(void* const* d_in, const int* in_sizes, int n_in,
                              void* d_out, int out_size, void* d_ws, size_t ws_size,
                              hipStream_t stream) {
  (void)in_sizes; (void)n_in; (void)ws_size;

  const float* x     = (const float*)d_in[0];
  const int*   src   = (const int*)d_in[1];
  const int*   dst   = (const int*)d_in[2];
  const int*   gid   = (const int*)d_in[3];
  const float* s1_ws = (const float*)d_in[4];
  const float* s1_wn = (const float*)d_in[5];
  const float* s1_b  = (const float*)d_in[6];
  const float* g2_w  = (const float*)d_in[7];
  const float* g2_b  = (const float*)d_in[8];
  const float* s3_ws = (const float*)d_in[9];
  const float* s3_wn = (const float*)d_in[10];
  const float* s3_b  = (const float*)d_in[11];
  const float* s4_ws = (const float*)d_in[12];
  const float* s4_wn = (const float*)d_in[13];
  const float* s4_b  = (const float*)d_in[14];
  const float* s5_ws = (const float*)d_in[15];
  const float* s5_wn = (const float*)d_in[16];
  const float* s5_b  = (const float*)d_in[17];
  const float* g6_w  = (const float*)d_in[18];
  const float* g6_b  = (const float*)d_in[19];
  const float* g7_w  = (const float*)d_in[20];
  const float* g7_b  = (const float*)d_in[21];
  float* out = (float*)d_out;

  // Workspace carve (~130 MB total)
  char* base = (char*)d_ws;
  size_t off = 0;
  auto carve = [&](size_t bytes) -> char* {
    char* p = base + off;
    off = (off + bytes + 255) & ~(size_t)255;
    return p;
  };
  f16*   ping   = (f16*)  carve((size_t)NNODES * LD * 2);
  f16*   pong   = (f16*)  carve((size_t)NNODES * LD * 2);
  float* aggf   = (float*)carve((size_t)NNODES * LD * 4);
  f16*   aggh   = (f16*)  carve((size_t)NNODES * LD * 2);
  float* inDeg  = (float*)carve((size_t)NNODES * 4);
  float* outDeg = (float*)carve((size_t)NNODES * 4);
  float* invIn  = (float*)carve((size_t)NNODES * 4);
  float* inIsq  = (float*)carve((size_t)NNODES * 4);
  float* outIsq = (float*)carve((size_t)NNODES * 4);
  float* cnt    = (float*)carve((size_t)NGRAPH * 4);
  f16* pW[11];
  for (int i = 0; i < 11; ++i) pW[i] = (f16*)carve((size_t)PACKED_HALVES * 2);

  const int TPB = 256;
  const int NELEM = NNODES * LD;
  const int DEG_BLK  = (NEDGES + TPB - 1) / TPB;
  const int NODE_BLK = (NNODES + TPB - 1) / TPB;
  const int CVT_BLK  = (NELEM + TPB - 1) / TPB;
  const int AGG_BLK  = (NEDGES * 32 + TPB - 1) / TPB;
  const int GEMM_BLK = (NNODES / 16) * NTILES / (TPB / 32);   // 6250, exact
  const int PACK_BLK = (PACKED_HALVES + TPB - 1) / TPB;

  // degrees + input conversion
  hipMemsetAsync(inDeg, 0, (size_t)NNODES * 4, stream);
  hipMemsetAsync(outDeg, 0, (size_t)NNODES * 4, stream);
  hipMemsetAsync(cnt, 0, (size_t)NGRAPH * 4, stream);
  hipMemsetAsync(d_out, 0, (size_t)out_size * 4, stream);
  deg_count<<<DEG_BLK, TPB, 0, stream>>>(src, dst, inDeg, outDeg);
  deg_final<<<NODE_BLK, TPB, 0, stream>>>(inDeg, outDeg, invIn, inIsq, outIsq);
  cvt_f32_f16<<<CVT_BLK, TPB, 0, stream>>>(x, ping, NELEM);

  // pack weights into WMMA B panels (zero-padded to 128x128)
  const float* wsrc[11] = { s1_ws, s1_wn, g2_w, s3_ws, s3_wn, s4_ws, s4_wn,
                            s5_ws, s5_wn, g6_w, g7_w };
  const int wK[11]   = { 128, 128, 128, 118, 118, 108, 108, 98, 98, 88, 83 };
  const int wN[11]   = { 128, 128, 118, 108, 108, 98, 98, 88, 88, 83, 5 };
  for (int i = 0; i < 11; ++i)
    pack_w<<<PACK_BLK, TPB, 0, stream>>>(wsrc[i], wK[i], wN[i], pW[i]);

  // L1 SAGE 128->128 : ping -> pong
  hipMemsetAsync(aggf, 0, (size_t)NELEM * 4, stream);
  agg_scatter<<<AGG_BLK, TPB, 0, stream>>>(ping, src, dst, nullptr, aggf);
  cvt_agg<<<CVT_BLK, TPB, 0, stream>>>(aggf, invIn, aggh);
  gemm_wmma<<<GEMM_BLK, TPB, 0, stream>>>(ping, pW[0], aggh, pW[1], s1_b, 128, 1, pong);

  // L2 GCN 128->118 : pong -> ping
  hipMemsetAsync(aggf, 0, (size_t)NELEM * 4, stream);
  agg_scatter<<<AGG_BLK, TPB, 0, stream>>>(pong, src, dst, outIsq, aggf);
  cvt_agg<<<CVT_BLK, TPB, 0, stream>>>(aggf, inIsq, aggh);
  gemm_wmma<<<GEMM_BLK, TPB, 0, stream>>>(aggh, pW[2], nullptr, nullptr, g2_b, 118, 1, ping);

  // L3 SAGE 118->108 : ping -> pong
  hipMemsetAsync(aggf, 0, (size_t)NELEM * 4, stream);
  agg_scatter<<<AGG_BLK, TPB, 0, stream>>>(ping, src, dst, nullptr, aggf);
  cvt_agg<<<CVT_BLK, TPB, 0, stream>>>(aggf, invIn, aggh);
  gemm_wmma<<<GEMM_BLK, TPB, 0, stream>>>(ping, pW[3], aggh, pW[4], s3_b, 108, 1, pong);

  // L4 SAGE 108->98 : pong -> ping
  hipMemsetAsync(aggf, 0, (size_t)NELEM * 4, stream);
  agg_scatter<<<AGG_BLK, TPB, 0, stream>>>(pong, src, dst, nullptr, aggf);
  cvt_agg<<<CVT_BLK, TPB, 0, stream>>>(aggf, invIn, aggh);
  gemm_wmma<<<GEMM_BLK, TPB, 0, stream>>>(pong, pW[5], aggh, pW[6], s4_b, 98, 1, ping);

  // L5 SAGE 98->88 : ping -> pong
  hipMemsetAsync(aggf, 0, (size_t)NELEM * 4, stream);
  agg_scatter<<<AGG_BLK, TPB, 0, stream>>>(ping, src, dst, nullptr, aggf);
  cvt_agg<<<CVT_BLK, TPB, 0, stream>>>(aggf, invIn, aggh);
  gemm_wmma<<<GEMM_BLK, TPB, 0, stream>>>(ping, pW[7], aggh, pW[8], s5_b, 88, 1, pong);

  // L6 GCN 88->83 : pong -> ping
  hipMemsetAsync(aggf, 0, (size_t)NELEM * 4, stream);
  agg_scatter<<<AGG_BLK, TPB, 0, stream>>>(pong, src, dst, outIsq, aggf);
  cvt_agg<<<CVT_BLK, TPB, 0, stream>>>(aggf, inIsq, aggh);
  gemm_wmma<<<GEMM_BLK, TPB, 0, stream>>>(aggh, pW[9], nullptr, nullptr, g6_b, 83, 1, ping);

  // L7 GCN 83->5 (no relu) : ping -> pong
  hipMemsetAsync(aggf, 0, (size_t)NELEM * 4, stream);
  agg_scatter<<<AGG_BLK, TPB, 0, stream>>>(ping, src, dst, outIsq, aggf);
  cvt_agg<<<CVT_BLK, TPB, 0, stream>>>(aggf, inIsq, aggh);
  gemm_wmma<<<GEMM_BLK, TPB, 0, stream>>>(aggh, pW[10], nullptr, nullptr, g7_b, 5, 0, pong);

  // per-graph mean pooling
  pool_sum<<<NODE_BLK, TPB, 0, stream>>>(pong, gid, out, cnt);
  pool_div<<<(NGRAPH * 5 + TPB - 1) / TPB, TPB, 0, stream>>>(out, cnt);
}
